// MHA_59072980189738
// MI455X (gfx1250) — matrix operand order repaired and checked by reading, not compiled, not run on hardware
//
#include <hip/hip_runtime.h>

// Problem constants (match reference)
#define BB 2
#define SS 2048
#define DD 1024
#define HH 16
#define DK 64

typedef _Float16 v16h __attribute__((ext_vector_type(16)));
typedef _Float16 v8h  __attribute__((ext_vector_type(8)));
typedef float    v8f  __attribute__((ext_vector_type(8)));

// ---------------------------------------------------------------------------
// Async global->LDS copy of 16 bytes (ASYNCcnt path) + wait.
// Inline asm (portable across toolchains; builtin pointer-AS types differ).
// VDST VGPR = LDS byte offset (hardware adds LDS_BASE); low 32 bits of a
// generic pointer to a __shared__ object are exactly that offset.
// ---------------------------------------------------------------------------
__device__ __forceinline__ void async_b128(const _Float16* g, _Float16* l) {
  unsigned lo = (unsigned)(size_t)l;
  asm volatile("global_load_async_to_lds_b128 %0, %1, off"
               :: "v"(lo), "v"(g) : "memory");
}

__device__ __forceinline__ void wait_async0() {
  asm volatile("s_wait_asynccnt 0x0" ::: "memory");
}

// ---------------------------------------------------------------------------
// Fragment loaders (layouts per CDNA5 ISA 7.12.2, wave32). 16B-aligned loads
// only, so LDS tiles may use odd (padded) row strides.
// A 16x32 f16: lane L<16 -> row L, K {0..7,16..23}; lane L+16 -> K {8..15,24..31}
// ---------------------------------------------------------------------------
__device__ __forceinline__ v16h load_a_frag(const _Float16* base, int ld) {
  int lane = threadIdx.x & 31;
  int l16  = lane & 15;
  int hi   = lane >> 4;
  const _Float16* p = base + l16 * ld + hi * 8;
  v8h lo = *(const v8h*)(p);
  v8h hg = *(const v8h*)(p + 16);
  return __builtin_shufflevector(lo, hg, 0,1,2,3,4,5,6,7,8,9,10,11,12,13,14,15);
}

// B 32x16 f16 from row-major src[N][K] (B[k,n] = src[n][k]):
// lane n<16 -> col n, K {0..7,16..23} ; lane n+16 -> col n, K {8..15,24..31}
__device__ __forceinline__ v16h load_b_frag(const _Float16* base, int ld) {
  int lane = threadIdx.x & 31;
  int l16  = lane & 15;
  int hi   = lane >> 4;
  const _Float16* p = base + l16 * ld + hi * 8;
  v8h lo = *(const v8h*)(p);
  v8h hg = *(const v8h*)(p + 16);
  return __builtin_shufflevector(lo, hg, 0,1,2,3,4,5,6,7,8,9,10,11,12,13,14,15);
}

__device__ __forceinline__ v8f wmma_f16(v16h a, v16h b, v8f c) {
  return __builtin_amdgcn_wmma_f32_16x16x32_f16(false, a, false, b,
                                                (short)0, c, false, false);
}

// ---------------------------------------------------------------------------
// f32 -> f16 conversion
// ---------------------------------------------------------------------------
__global__ void cvt_f32_f16(const float* __restrict__ in, _Float16* __restrict__ out, int n) {
  int i = blockIdx.x * blockDim.x + threadIdx.x;
  if (i < n) out[i] = (_Float16)in[i];
}

// ---------------------------------------------------------------------------
// V transpose: Vt[((b*H+h)*DK+dk)*S + t] = Vp[(b*S+t)*D + h*DK + dk]
// ---------------------------------------------------------------------------
__global__ void transpose_v(const _Float16* __restrict__ Vp, _Float16* __restrict__ Vt) {
  int i = blockIdx.x * blockDim.x + threadIdx.x;   // over B*H*DK*S
  if (i >= BB * HH * DK * SS) return;
  int t    = i % SS;
  int rest = i / SS;
  int dk   = rest % DK;
  int bh   = rest / DK;
  int b    = bh / HH;
  int h    = bh % HH;
  Vt[i] = Vp[(size_t)(b * SS + t) * DD + h * DK + dk];
}

// ---------------------------------------------------------------------------
// GEMM: Y[M][N] = A[M][K] @ W[N][K]^T + bias.
// Block = 8 waves: 8 consecutive 16-row M tiles x one 64-wide N supertile.
// W k-slab (64x32 f16) async-staged to LDS, double buffered, shared by all
// 8 waves. One s_wait_asynccnt + one barrier per 32-wide k-step.
// ---------------------------------------------------------------------------
#define WLD 40   // padded LDS row stride (halfs) for 32-half rows

__global__ void gemm_rt(const _Float16* __restrict__ A, const _Float16* __restrict__ W,
                        const float* __restrict__ bias,
                        _Float16* __restrict__ out16, float* __restrict__ out32,
                        int M, int N, int K) {
  __shared__ _Float16 Wl[2][64 * WLD];

  int tid   = threadIdx.x;
  int wslot = tid >> 5;
  int ntn   = N >> 6;
  int nt    = blockIdx.x % ntn;
  int mt    = (blockIdx.x / ntn) * 8 + wslot;

  // cooperative loader mapping: 64 rows x 4 segments of 8 halfs
  int lrow = tid >> 2;
  int lseg = tid & 3;
  const _Float16* gW = W + (size_t)(nt * 64 + lrow) * K + lseg * 8;
  _Float16* lW0 = &Wl[0][lrow * WLD + lseg * 8];
  _Float16* lW1 = &Wl[1][lrow * WLD + lseg * 8];

  async_b128(gW, lW0);                      // prologue: slab 0

  v8f acc[4];
  #pragma unroll
  for (int c = 0; c < 4; ++c) acc[c] = (v8f){0.f,0.f,0.f,0.f,0.f,0.f,0.f,0.f};

  const _Float16* Ab = A + (size_t)mt * 16 * K;
  int nk = K >> 5;
  for (int i = 0; i < nk; ++i) {
    wait_async0();
    __syncthreads();                        // slab i visible; slab i-1 consumers done
    if (i + 1 < nk)
      async_b128(gW + (size_t)(i + 1) * 32, ((i + 1) & 1) ? lW1 : lW0);
    const _Float16* Wb = &Wl[i & 1][0];
    v16h a = load_a_frag(Ab + i * 32, K);
    #pragma unroll
    for (int c = 0; c < 4; ++c)
      acc[c] = wmma_f16(a, load_b_frag(Wb + (c * 16) * WLD, WLD), acc[c]);
  }

  int lane = tid & 31;
  int l16  = lane & 15;
  int r8   = (lane >> 4) * 8;
  #pragma unroll
  for (int c = 0; c < 4; ++c) {
    int col = nt * 64 + c * 16 + l16;
    float bv = bias ? bias[col] : 0.f;
    #pragma unroll
    for (int r = 0; r < 8; ++r) {
      int row = mt * 16 + r8 + r;
      float v = acc[c][r] + bv;
      if (out32) out32[(size_t)row * N + col] = v;
      else       out16[(size_t)row * N + col] = (_Float16)v;
    }
  }
}

// ---------------------------------------------------------------------------
// Flash attention (causal). Block = 8 waves = 128 query rows of one (b,h).
// Per 32-key block: K tile (32x64) and V^T tile (64x32) async-staged to LDS
// (double buffered), consumed by all 8 waves. Q stays in registers.
// Fully-masked tail blocks for low-q waves are benign (alpha=1, p~=0).
// ---------------------------------------------------------------------------
#define KLD 72   // padded LDS row stride (halfs) for 64-half K rows
#define PLD 40   // padded LDS row stride for P / V tiles

__global__ void attn_kernel(const _Float16* __restrict__ Q, const _Float16* __restrict__ K,
                            const _Float16* __restrict__ Vt, _Float16* __restrict__ O) {
  __shared__ _Float16 Kl[2][32 * KLD];
  __shared__ _Float16 Vl[2][64 * PLD];
  __shared__ _Float16 Pl[8][16 * PLD];

  int tid   = threadIdx.x;
  int wslot = tid >> 5;
  int bh    = blockIdx.x >> 4;          // (b,h)
  int chunk = blockIdx.x & 15;          // 128-row query chunk
  int b     = bh >> 4;
  int h     = bh & 15;
  int qbase = chunk * 128;
  int q0    = qbase + wslot * 16;
  int hoff  = h * DK;

  int lane = tid & 31;
  int l16  = lane & 15;
  int r8   = (lane >> 4) * 8;
  int col  = l16;

  // cooperative loader mappings
  int krow = tid >> 3, kseg = tid & 7;  // K tile: 32 rows x 8 segs of 8 halfs
  int vrow = tid >> 2, vseg = tid & 3;  // V tile: 64 rows x 4 segs of 8 halfs
  const _Float16* gK = K + (size_t)b * SS * DD + (size_t)krow * DD + hoff + kseg * 8;
  const _Float16* gV = Vt + ((size_t)bh * DK + vrow) * SS + vseg * 8;
  _Float16* lK = &Kl[0][krow * KLD + kseg * 8];
  _Float16* lV = &Vl[0][vrow * PLD + vseg * 8];
  int kbuf = 32 * KLD;                  // buffer stride (halfs)
  int vbuf = 64 * PLD;

  // prologue: key block 0
  async_b128(gK, lK);
  async_b128(gV, lV);

  const _Float16* Qb = Q + (size_t)(b * SS + q0) * DD + hoff;
  v16h aq0 = load_a_frag(Qb, DD);
  v16h aq1 = load_a_frag(Qb + 32, DD);

  float mr[8], lr[8];
  #pragma unroll
  for (int r = 0; r < 8; ++r) { mr[r] = -1e30f; lr[r] = 0.f; }
  v8f o0 = (v8f){0.f,0.f,0.f,0.f,0.f,0.f,0.f,0.f};
  v8f o1 = o0, o2 = o0, o3 = o0;

  _Float16* Pb = &Pl[wslot][0];
  const float sc = 0.125f;              // 1/sqrt(DK)

  int njj = chunk * 4 + 4;              // 32-key blocks covering rows <= qbase+127
  for (int jj = 0; jj < njj; ++jj) {
    wait_async0();
    __syncthreads();                    // tiles for block jj resident
    if (jj + 1 < njj) {
      size_t t1 = (size_t)(jj + 1) * 32;
      int nb = (jj + 1) & 1;
      async_b128(gK + t1 * DD, lK + nb * kbuf);
      async_b128(gV + t1,      lV + nb * vbuf);
    }
    int bb = jj & 1;
    const _Float16* Kt = &Kl[bb][0];
    const _Float16* Vb = &Vl[bb][0];
    int t0 = jj * 32;

    v8f s0 = (v8f){0.f,0.f,0.f,0.f,0.f,0.f,0.f,0.f};
    v8f s1 = s0;
    s0 = wmma_f16(aq0, load_b_frag(Kt,                 KLD), s0);
    s0 = wmma_f16(aq1, load_b_frag(Kt + 32,            KLD), s0);
    s1 = wmma_f16(aq0, load_b_frag(Kt + 16 * KLD,      KLD), s1);
    s1 = wmma_f16(aq1, load_b_frag(Kt + 16 * KLD + 32, KLD), s1);

    float alpha[8];
    #pragma unroll
    for (int r = 0; r < 8; ++r) {
      int gr = q0 + r8 + r;
      float v0 = s0[r] * sc; if (t0 + col      > gr) v0 = -1000.f;
      float v1 = s1[r] * sc; if (t0 + 16 + col > gr) v1 = -1000.f;
      float mx = fmaxf(v0, v1);
      #pragma unroll
      for (int off = 8; off; off >>= 1) mx = fmaxf(mx, __shfl_xor(mx, off, 16));
      float mn = fmaxf(mr[r], mx);
      float a  = __expf(mr[r] - mn);
      float p0 = __expf(v0 - mn);
      float p1 = __expf(v1 - mn);
      float rs = p0 + p1;
      #pragma unroll
      for (int off = 8; off; off >>= 1) rs += __shfl_xor(rs, off, 16);
      lr[r] = lr[r] * a + rs;
      mr[r] = mn;
      alpha[r] = a;
      Pb[(r8 + r) * PLD + col]      = (_Float16)p0;
      Pb[(r8 + r) * PLD + 16 + col] = (_Float16)p1;
    }
    #pragma unroll
    for (int r = 0; r < 8; ++r) {
      o0[r] *= alpha[r]; o1[r] *= alpha[r]; o2[r] *= alpha[r]; o3[r] *= alpha[r];
    }
    v16h ap = load_a_frag(Pb, PLD);     // P (16x32) as A fragment, via LDS
    o0 = wmma_f16(ap, load_b_frag(Vb,            PLD), o0);
    o1 = wmma_f16(ap, load_b_frag(Vb + 16 * PLD, PLD), o1);
    o2 = wmma_f16(ap, load_b_frag(Vb + 32 * PLD, PLD), o2);
    o3 = wmma_f16(ap, load_b_frag(Vb + 48 * PLD, PLD), o3);
  }

  _Float16* Ob = O + (size_t)(b * SS + q0) * DD + hoff;
  #pragma unroll
  for (int r = 0; r < 8; ++r) {
    float inv = 1.0f / lr[r];
    int row = r8 + r;
    Ob[(size_t)row * DD + col]      = (_Float16)(o0[r] * inv);
    Ob[(size_t)row * DD + 16 + col] = (_Float16)(o1[r] * inv);
    Ob[(size_t)row * DD + 32 + col] = (_Float16)(o2[r] * inv);
    Ob[(size_t)row * DD + 48 + col] = (_Float16)(o3[r] * inv);
  }
}

// ---------------------------------------------------------------------------
// Host launcher
// ---------------------------------------------------------------------------
extern "C" void kernel_launch(void* const* d_in, const int* in_sizes, int n_in,
                              void* d_out, int out_size, void* d_ws, size_t ws_size,
                              hipStream_t stream) {
  (void)in_sizes; (void)n_in; (void)out_size; (void)ws_size;
  const float* q  = (const float*)d_in[0];
  const float* k  = (const float*)d_in[1];
  const float* v  = (const float*)d_in[2];
  const float* Wq = (const float*)d_in[3];
  const float* bq = (const float*)d_in[4];
  const float* Wk = (const float*)d_in[5];
  const float* bk = (const float*)d_in[6];
  const float* Wv = (const float*)d_in[7];
  const float* bv = (const float*)d_in[8];
  const float* Wo = (const float*)d_in[9];
  const float* bo = (const float*)d_in[10];
  // d_in[11] = mask: causal, handled analytically.

  const int NTOK = BB * SS;            // 4096
  const int NX   = NTOK * DD;          // 4,194,304
  const int NW   = DD * DD;            // 1,048,576

  _Float16* w = (_Float16*)d_ws;
  _Float16* q16  = w; w += NX;
  _Float16* k16  = w; w += NX;
  _Float16* v16  = w; w += NX;
  _Float16* Wq16 = w; w += NW;
  _Float16* Wk16 = w; w += NW;
  _Float16* Wv16 = w; w += NW;
  _Float16* Wo16 = w; w += NW;
  _Float16* Qp   = w; w += NX;
  _Float16* Kp   = w; w += NX;
  _Float16* Vp   = w; w += NX;
  _Float16* Vt   = w; w += NX;
  _Float16* Ao   = w; w += NX;

  const int TPB = 256;
  // 1) f32 -> f16 conversions
  cvt_f32_f16<<<(NX + TPB - 1) / TPB, TPB, 0, stream>>>(q,  q16, NX);
  cvt_f32_f16<<<(NX + TPB - 1) / TPB, TPB, 0, stream>>>(k,  k16, NX);
  cvt_f32_f16<<<(NX + TPB - 1) / TPB, TPB, 0, stream>>>(v,  v16, NX);
  cvt_f32_f16<<<(NW + TPB - 1) / TPB, TPB, 0, stream>>>(Wq, Wq16, NW);
  cvt_f32_f16<<<(NW + TPB - 1) / TPB, TPB, 0, stream>>>(Wk, Wk16, NW);
  cvt_f32_f16<<<(NW + TPB - 1) / TPB, TPB, 0, stream>>>(Wv, Wv16, NW);
  cvt_f32_f16<<<(NW + TPB - 1) / TPB, TPB, 0, stream>>>(Wo, Wo16, NW);

  // 2) QKV projections: (M/16/8)*(N/64) blocks of 8 waves
  int gemm_blocks = (NTOK / 16 / 8) * (DD / 64);   // 512
  gemm_rt<<<gemm_blocks, TPB, 0, stream>>>(q16, Wq16, bq, Qp, nullptr, NTOK, DD, DD);
  gemm_rt<<<gemm_blocks, TPB, 0, stream>>>(k16, Wk16, bk, Kp, nullptr, NTOK, DD, DD);
  gemm_rt<<<gemm_blocks, TPB, 0, stream>>>(v16, Wv16, bv, Vp, nullptr, NTOK, DD, DD);

  // 3) V transpose to [B*H*DK][S]
  transpose_v<<<(NX + TPB - 1) / TPB, TPB, 0, stream>>>(Vp, Vt);

  // 4) Flash attention: (B*H) x (S/128) blocks
  attn_kernel<<<BB * HH * (SS / 128), TPB, 0, stream>>>(Qp, Kp, Vt, Ao);

  // 5) Output projection, f32 result straight into d_out
  gemm_rt<<<gemm_blocks, TPB, 0, stream>>>(Ao, Wo16, bo, nullptr, (float*)d_out, NTOK, DD, DD);
}
